// GATLayer_28089086116641
// MI455X (gfx1250) — matrix operand order repaired and checked by reading, not compiled, not run on hardware
//
#include <hip/hip_runtime.h>

// ---------------------------------------------------------------------------
// GAT layer, fused flash-attention style, for MI455X (gfx1250, wave32, WMMA)
// N=4096, F_IN=512, H=4, D=64
// ---------------------------------------------------------------------------

typedef __attribute__((ext_vector_type(16))) _Float16 v16h;
typedef __attribute__((ext_vector_type(8)))  _Float16 v8h;
typedef __attribute__((ext_vector_type(8)))  float    v8f;
typedef __attribute__((ext_vector_type(4)))  float    v4f;
typedef __attribute__((ext_vector_type(4)))  int      v4i;

#define GAT_N    4096
#define GAT_FIN  512
#define GAT_H    4
#define GAT_D    64
#define GAT_HD   256
#define GAT_ALPHA 0.2f
#define GAT_NEG  (-9.0e15f)

// ---------------------------------------------------------------------------
// Stage 1a: x (f32, N x FIN) -> f16 row-major (A operand source)
// ---------------------------------------------------------------------------
__global__ __launch_bounds__(256) void cvt_x_f16(const float* __restrict__ x,
                                                 _Float16* __restrict__ xh) {
  int i = blockIdx.x * blockDim.x + threadIdx.x;
  if (i < GAT_N * GAT_FIN) xh[i] = (_Float16)x[i];
}

// ---------------------------------------------------------------------------
// Stage 1b: W (f32, FIN x HD) -> f16 transposed WT[n][k] (B operand source)
// ---------------------------------------------------------------------------
__global__ __launch_bounds__(256) void cvt_w_f16T(const float* __restrict__ W,
                                                  _Float16* __restrict__ wT) {
  int i = blockIdx.x * blockDim.x + threadIdx.x;
  if (i >= GAT_FIN * GAT_HD) return;
  int k = i / GAT_HD;
  int n = i % GAT_HD;
  wT[n * GAT_FIN + k] = (_Float16)W[i];
}

// ---------------------------------------------------------------------------
// Stage 2: h = x @ W via v_wmma_f32_16x16x32_f16.
// One wave per (16-row, head) tile; 4 accumulators cover the head's 64 cols.
// Writes h (f32, N x 256) and hT (f16, [head*64+d][n]) for stage 4's B operand.
// ---------------------------------------------------------------------------
__global__ __launch_bounds__(256) void gemm_h(const _Float16* __restrict__ xh,
                                              const _Float16* __restrict__ wT,
                                              float* __restrict__ hout,
                                              _Float16* __restrict__ hT) {
  const int lane   = threadIdx.x & 31;
  const int wid    = blockIdx.x * (blockDim.x >> 5) + (threadIdx.x >> 5);
  const int head   = wid & 3;
  const int i0     = (wid >> 2) * 16;
  const int mrow   = lane & 15;
  const int hiHalf = lane >> 4;

  v8f acc[4] = {};
  const _Float16* arow = xh + (size_t)(i0 + mrow) * GAT_FIN;
  const int aoff = hiHalf * 8;

  for (int k0 = 0; k0 < GAT_FIN; k0 += 32) {
    // A fragment: lane<16 holds K 0..15, lane>=16 holds K 16..31 (contiguous,
    // matching the B fragment).
    v16h a = *(const v16h*)(arow + k0 + aoff * 2);

#pragma unroll
    for (int dt = 0; dt < 4; ++dt) {
      // B fragment: lane holds N = mrow, K = k0 + hiHalf*16 .. +15 contiguous
      const _Float16* bp =
          wT + (size_t)(head * 64 + dt * 16 + mrow) * GAT_FIN + k0 + hiHalf * 16;
      v16h b = *(const v16h*)bp;
      acc[dt] = __builtin_amdgcn_wmma_f32_16x16x32_f16(
          false, a, false, b, (short)0, acc[dt], false, false);
    }
  }

  // C layout: VGPR r -> (M = r + hiHalf*8, N = mrow) per 16x16 f32 tile
#pragma unroll
  for (int dt = 0; dt < 4; ++dt) {
#pragma unroll
    for (int r = 0; r < 8; ++r) {
      int m   = r + hiHalf * 8;
      int col = head * 64 + dt * 16 + mrow;
      float v = acc[dt][r];
      hout[(size_t)(i0 + m) * GAT_HD + col] = v;
      hT[(size_t)col * GAT_N + (i0 + m)]    = (_Float16)v;
    }
  }
}

// ---------------------------------------------------------------------------
// Stage 3: s_self[h][n], s_neigh[h][n] = <h[n,h,:], a_*[h,:]>  (tiny)
// ---------------------------------------------------------------------------
__global__ __launch_bounds__(256) void sdot(const float* __restrict__ hmat,
                                            const float* __restrict__ a_self,
                                            const float* __restrict__ a_neigh,
                                            float* __restrict__ ss,
                                            float* __restrict__ sn) {
  int t = blockIdx.x * blockDim.x + threadIdx.x;
  if (t >= GAT_H * GAT_N) return;
  int hh = t >> 12;          // N = 4096
  int n  = t & (GAT_N - 1);
  const float* hp = hmat + (size_t)n * GAT_HD + hh * GAT_D;
  const float* as = a_self + hh * GAT_D;
  const float* an = a_neigh + hh * GAT_D;
  float s0 = 0.f, s1 = 0.f;
#pragma unroll 8
  for (int d = 0; d < GAT_D; ++d) {
    float hv = hp[d];
    s0 += hv * as[d];
    s1 += hv * an[d];
  }
  ss[t] = s0;
  sn[t] = s1;
}

// ---------------------------------------------------------------------------
// Stage 4: fused scores + LeakyReLU + mask + online softmax + attn@h.
// 1 wave per (head, 16-row tile); K loop over j in steps of 32 (WMMA K).
// adj and M are streamed exactly once (dominant 128 MB of traffic) with
// global_prefetch_b8 ahead of the stream. The shared hT B-tile (4 KB) is
// double-buffered through LDS (one split-barrier per iteration) and the
// s_neigh slab (16 KB) is staged once per block — cutting ~750 MB of
// redundant L2 reads down to ~130 MB.
// ---------------------------------------------------------------------------
__global__ __launch_bounds__(128) void attn_fused(const int* __restrict__ adj,
                                                  const float* __restrict__ Mw,
                                                  const _Float16* __restrict__ hT,
                                                  const float* __restrict__ ss,
                                                  const float* __restrict__ sn,
                                                  float* __restrict__ out) {
  __shared__ _Float16 lds_h[2][GAT_D * 32];   // 2 x 4 KB double buffer
  __shared__ float    lds_sn[GAT_N];          // 16 KB s_neigh slab

  const int tid    = threadIdx.x;
  const int lane   = tid & 31;
  const int w      = tid >> 5;
  const int hh     = blockIdx.x >> 6;          // N/64 = 64 row-blocks
  const int ib     = blockIdx.x & 63;
  const int i0     = ib * 64 + w * 16;
  const int mrow   = lane & 15;
  const int hiHalf = lane >> 4;
  const int row    = i0 + mrow;

  const float* snh    = sn + hh * GAT_N;
  const _Float16* hTh = hT + (size_t)hh * GAT_D * GAT_N;

  // --- stage s_neigh slab (each of 128 threads copies 32 floats) ---
  {
    const int base = tid * 32;
#pragma unroll
    for (int c = 0; c < 8; ++c)
      *(v4f*)&lds_sn[base + c * 4] = *(const v4f*)(snh + base + c * 4);
  }
  // --- stage hT tile 0 (each thread copies one 32B chunk) ---
  const int sd = tid >> 1;   // d row 0..63
  const int sh = tid & 1;    // 16-half chunk within the 32-j tile
  {
    v16h t0 = *(const v16h*)(hTh + (size_t)sd * GAT_N + sh * 16);
    *(v16h*)&lds_h[0][sd * 32 + sh * 16] = t0;
  }
  __syncthreads();

  const float s_self_row = ss[hh * GAT_N + row];
  const float* Mrow = Mw + (size_t)row * GAT_N;
  const int*   Arow = adj + (size_t)row * GAT_N;

  v8f acc[4] = {};
  float m_run = -3.0e38f;
  float l_run = 0.0f;

  for (int jt = 0; jt < GAT_N / 32; ++jt) {
    const int j0  = jt * 32;
    const int cur = jt & 1;
    const bool have_nxt = (jt + 1) < (GAT_N / 32);

    // issue next hT tile's global load early; VALU below hides the latency
    v16h nxt;
    if (have_nxt)
      nxt = *(const v16h*)(hTh + (size_t)sd * GAT_N + (j0 + 32) + sh * 16);

    const int jb = j0 + hiHalf * 8;  // this lane's A-fragment K slots

    v4f m0 = *(const v4f*)(Mrow + jb);
    v4f m1 = *(const v4f*)(Mrow + jb + 4);
    v4f m2 = *(const v4f*)(Mrow + jb + 16);
    v4f m3 = *(const v4f*)(Mrow + jb + 20);
    v4i a0 = *(const v4i*)(Arow + jb);
    v4i a1 = *(const v4i*)(Arow + jb + 4);
    v4i a2 = *(const v4i*)(Arow + jb + 16);
    v4i a3 = *(const v4i*)(Arow + jb + 20);
    v4f n0 = *(const v4f*)&lds_sn[jb];
    v4f n1 = *(const v4f*)&lds_sn[jb + 4];
    v4f n2 = *(const v4f*)&lds_sn[jb + 16];
    v4f n3 = *(const v4f*)&lds_sn[jb + 20];

    if (have_nxt) {   // stream-prefetch next M/adj tile (global_prefetch_b8)
      __builtin_prefetch(Mrow + jb + 32, 0, 1);
      __builtin_prefetch(Arow + jb + 32, 0, 1);
    }

    float Mv[16], Sv[16];
    int   Av[16];
#pragma unroll
    for (int e = 0; e < 4; ++e) {
      Mv[e] = m0[e];      Mv[4 + e] = m1[e];
      Mv[8 + e] = m2[e];  Mv[12 + e] = m3[e];
      Av[e] = a0[e];      Av[4 + e] = a1[e];
      Av[8 + e] = a2[e];  Av[12 + e] = a3[e];
      Sv[e] = n0[e];      Sv[4 + e] = n1[e];
      Sv[8 + e] = n2[e];  Sv[12 + e] = n3[e];
    }

    float sc[16];
    float tm = -3.0e38f;
#pragma unroll
    for (int e = 0; e < 16; ++e) {
      float s = (s_self_row + Sv[e]) * Mv[e];
      s = (s > 0.f) ? s : GAT_ALPHA * s;      // LeakyReLU
      s = (Av[e] > 0) ? s : GAT_NEG;          // edge mask
      sc[e] = s;
      tm = fmaxf(tm, s);
    }
    // row max lives split across lane pair (L, L+16)
    tm = fmaxf(tm, __shfl_xor(tm, 16, 32));
    float m_new   = fmaxf(m_run, tm);
    float rescale = __expf(m_run - m_new);

    float psum = 0.f;
    _Float16 pv[16];
#pragma unroll
    for (int e = 0; e < 16; ++e) {
      float p = __expf(sc[e] - m_new);
      psum += p;
      pv[e] = (_Float16)p;
    }
    psum += __shfl_xor(psum, 16, 32);
    l_run = l_run * rescale + psum;
    m_run = m_new;

    // rescale accumulators: C VGPR r holds row (r + hiHalf*8); stats for row m
    // are replicated at lanes m and m+16, so broadcast from lane m.
#pragma unroll
    for (int r = 0; r < 8; ++r) {
      float f = __shfl(rescale, r + hiHalf * 8, 32);
      acc[0][r] *= f;
      acc[1][r] *= f;
      acc[2][r] *= f;
      acc[3][r] *= f;
    }

    // A fragment from probabilities (layout matches score slots by design)
    v16h afrag;
#pragma unroll
    for (int e = 0; e < 16; ++e) afrag[e] = pv[e];

    // B fragments from the LDS double buffer (ds_load_b128)
#pragma unroll
    for (int dt = 0; dt < 4; ++dt) {
      v8h blo = *(const v8h*)&lds_h[cur][(dt * 16 + mrow) * 32 + hiHalf * 8];
      v8h bhi = *(const v8h*)&lds_h[cur][(dt * 16 + mrow) * 32 + hiHalf * 8 + 16];
      v16h b;
#pragma unroll
      for (int e = 0; e < 8; ++e) { b[e] = blo[e]; b[e + 8] = bhi[e]; }
      acc[dt] = __builtin_amdgcn_wmma_f32_16x16x32_f16(
          false, afrag, false, b, (short)0, acc[dt], false, false);
    }

    // commit next tile to the other buffer; one barrier flips buffers
    if (have_nxt)
      *(v16h*)&lds_h[1 - cur][sd * 32 + sh * 16] = nxt;
    __syncthreads();
  }

  // normalize and write h_prime laid out (H, N, D) flattened
#pragma unroll
  for (int r = 0; r < 8; ++r) {
    int m   = r + hiHalf * 8;
    float l = __shfl(l_run, m, 32);
    float inv = (l > 0.f) ? (1.0f / l) : 0.0f;
#pragma unroll
    for (int dt = 0; dt < 4; ++dt) {
      int d = dt * 16 + mrow;
      out[(size_t)hh * GAT_N * GAT_D + (size_t)(i0 + m) * GAT_D + d] =
          acc[dt][r] * inv;
    }
  }
}

// ---------------------------------------------------------------------------
// Launch
// ---------------------------------------------------------------------------
extern "C" void kernel_launch(void* const* d_in, const int* in_sizes, int n_in,
                              void* d_out, int out_size, void* d_ws, size_t ws_size,
                              hipStream_t stream) {
  const float* x       = (const float*)d_in[0];
  const int*   adj     = (const int*)d_in[1];
  const float* Mw      = (const float*)d_in[2];
  const float* W       = (const float*)d_in[3];
  const float* a_self  = (const float*)d_in[4];
  const float* a_neigh = (const float*)d_in[5];
  float* out = (float*)d_out;

  char* ws = (char*)d_ws;
  size_t off = 0;
  _Float16* x_h  = (_Float16*)(ws + off); off += (size_t)GAT_N * GAT_FIN * 2;   // 4 MiB
  _Float16* wT   = (_Float16*)(ws + off); off += (size_t)GAT_HD * GAT_FIN * 2;  // 256 KiB
  float*    hmat = (float*)(ws + off);    off += (size_t)GAT_N * GAT_HD * 4;    // 4 MiB
  _Float16* hT   = (_Float16*)(ws + off); off += (size_t)GAT_HD * GAT_N * 2;    // 2 MiB
  float*    ssb  = (float*)(ws + off);    off += (size_t)GAT_H * GAT_N * 4;     // 64 KiB
  float*    snb  = (float*)(ws + off);    off += (size_t)GAT_H * GAT_N * 4;     // 64 KiB
  (void)ws_size; (void)in_sizes; (void)n_in; (void)out_size;

  // Stage 1: precision conversion
  cvt_x_f16<<<(GAT_N * GAT_FIN + 255) / 256, 256, 0, stream>>>(x, x_h);
  cvt_w_f16T<<<(GAT_FIN * GAT_HD + 255) / 256, 256, 0, stream>>>(W, wT);

  // Stage 2: h = x @ W  (1024 wave-tiles, 8 waves/block)
  gemm_h<<<(GAT_N / 16) * GAT_H / 8, 256, 0, stream>>>(x_h, wT, hmat, hT);

  // Stage 3: attention-vector dots
  sdot<<<(GAT_H * GAT_N + 255) / 256, 256, 0, stream>>>(hmat, a_self, a_neigh, ssb, snb);

  // Stage 4: fused masked-softmax attention (H * N/64 blocks, 4 waves each)
  attn_fused<<<GAT_H * (GAT_N / 64), 128, 0, stream>>>(adj, Mw, hT, ssb, snb, out);
}
